// EdgeConvGNNClassifier_18614388261117
// MI455X (gfx1250) — compile-verified, matrix operand order checked by reading, not run on hardware
//
#include <hip/hip_runtime.h>
#include <hip/hip_bf16.h>

typedef __attribute__((ext_vector_type(16))) _Float16 v16h;
typedef __attribute__((ext_vector_type(8)))  _Float16 v8h;
typedef __attribute__((ext_vector_type(2)))  _Float16 v2h;
typedef __attribute__((ext_vector_type(8)))  float    v8f;

union HFrag { v16h v; v2h p[8]; };

#define STR 40   // 32 halfs + 8 pad (keeps 16B alignment: 40*2 = 80 bytes)

// ---------------------------------------------------------------------------
// CDNA5 async global->LDS copy (16 bytes / lane), tracked with ASYNCcnt.
// VDST VGPR carries the LDS byte offset (flat aperture: addr[31:0] == LDS off).
// ---------------------------------------------------------------------------
__device__ __forceinline__ void async_copy_b128(const void* gsrc, void* lds_dst)
{
    unsigned lds = (unsigned)(unsigned long long)(uintptr_t)lds_dst;
    asm volatile("global_load_async_to_lds_b128 %0, %1, off"
                 :: "v"(lds), "v"(gsrc) : "memory");
}
__device__ __forceinline__ void wait_async0()
{
    asm volatile("s_wait_asynccnt 0" ::: "memory");
}
// Wave-local LDS ordering: wait for this wave's outstanding DS ops.
__device__ __forceinline__ void wait_ds0()
{
    asm volatile("s_wait_dscnt 0" ::: "memory");
}

// Build a 16x32 f16 WMMA A/B fragment from an LDS tile (row-major, stride STR).
// Per ISA 7.12.2: lane = hr + 16*hs; VGPR v -> K = (v&3)*2 + (v>=4)*16 + hs*8.
__device__ __forceinline__ v16h load_frag(const _Float16* base, int hr, int hs)
{
    HFrag f;
    const _Float16* rowp = base + hr * STR;
#pragma unroll
    for (int v = 0; v < 8; ++v) {
        int k = ((v & 3) << 1) + ((v >> 2) << 4) + (hs << 3);
        f.p[v] = *(const v2h*)(rowp + k);
    }
    return f.v;
}

// ---------------------------------------------------------------------------
// WMMA GEMM: C = act(A(MxK,f16) @ Bt^T + bias), Bt is N x K row-major
// (pre-transposed weights). Block tile 128(M) x 256(N) x 32(K), 256 threads
// (8 waves = 2x4), wave tile 64x64 (16 wmma / 8 fragment loads per K-step).
// Double-buffered LDS, async global->LDS staging, ONE barrier per K-step.
// ---------------------------------------------------------------------------
template<int RELU, int HASF, int HASH>
__global__ __launch_bounds__(256)
void wmma_gemm(const _Float16* __restrict__ A, const _Float16* __restrict__ Bt,
               const float* __restrict__ bias,
               float* __restrict__ Cf, _Float16* __restrict__ Ch,
               int M, int N, int K)
{
    __shared__ _Float16 sA[2][128 * STR];
    __shared__ _Float16 sB[2][256 * STR];

    const int t = threadIdx.x;
    const int lane = t & 31;
    const int w = t >> 5;
    const int hr = lane & 15, hs = lane >> 4;
    const int bm = blockIdx.y * 128;
    const int bn = blockIdx.x * 256;
    const int wm = (w & 1) * 64;     // 2 waves along M
    const int wn = (w >> 1) * 64;    // 4 waves along N
    const int lrow = t >> 2;         // 0..63, 4 threads/row
    const int lcol = (t & 3) * 8;    // 0,8,16,24

    auto stage = [&](int buf, int k0) {
#pragma unroll
        for (int p = 0; p < 2; ++p) {
            int r = lrow + p * 64;
            async_copy_b128(A + (size_t)(bm + r) * K + k0 + lcol, &sA[buf][r * STR + lcol]);
        }
#pragma unroll
        for (int p = 0; p < 4; ++p) {
            int r = lrow + p * 64;
            async_copy_b128(Bt + (size_t)(bn + r) * K + k0 + lcol, &sB[buf][r * STR + lcol]);
        }
    };

    v8f acc[4][4];
#pragma unroll
    for (int i = 0; i < 4; ++i)
#pragma unroll
        for (int j = 0; j < 4; ++j) acc[i][j] = (v8f)0.0f;

    const int nIter = K >> 5;
    stage(0, 0);
    for (int it = 0; it < nIter; ++it) {
        // Single barrier per K-step: tiles for `buf` are complete (async waited
        // by every wave) and all reads of `buf^1` finished before the previous
        // barrier, so staging buf^1 below is race-free.
        wait_async0();
        __syncthreads();
        if (it + 1 < nIter) stage((it + 1) & 1, (it + 1) << 5);

        const int buf = it & 1;
        v16h afrag[4], bfrag[4];
#pragma unroll
        for (int mi = 0; mi < 4; ++mi)
            afrag[mi] = load_frag(&sA[buf][(wm + mi * 16) * STR], hr, hs);
#pragma unroll
        for (int ni = 0; ni < 4; ++ni)
            bfrag[ni] = load_frag(&sB[buf][(wn + ni * 16) * STR], hr, hs);

#pragma unroll
        for (int mi = 0; mi < 4; ++mi)
#pragma unroll
            for (int ni = 0; ni < 4; ++ni)
                acc[mi][ni] = __builtin_amdgcn_wmma_f32_16x16x32_f16(
                    false, afrag[mi], false, bfrag[ni],
                    (short)0, acc[mi][ni], false, false);
    }

    // Epilogue (branch-free): v8f element v <-> (M = v + 8*hs, N = hr).
#pragma unroll
    for (int mi = 0; mi < 4; ++mi)
#pragma unroll
        for (int ni = 0; ni < 4; ++ni) {
            const int col  = bn + wn + ni * 16 + hr;
            const int row0 = bm + wm + mi * 16 + hs * 8;
            const float bsc = bias[col];
            size_t base = (size_t)row0 * N + col;
            float*    pf = Cf + base;
            _Float16* ph = Ch + base;
#pragma unroll
            for (int v = 0; v < 8; ++v) {
                float x = acc[mi][ni][v] + bsc;
                if (RELU) x = fmaxf(x, 0.0f);
                if (HASF) { *pf = x;            pf += N; }
                if (HASH) { *ph = (_Float16)x;  ph += N; }
            }
        }
}

// ---------------------------------------------------------------------------
// WMMA kNN (k=8): block = 128 queries; candidates streamed in 128-row panels.
// GEMM-shaped wave work: wave w owns query tiles {2*(w&3), 2*(w&3)+1} and
// candidate half (w>>2) -> 2 afrags x 4 bfrags -> 8 wmma from 6 fragment
// loads (multi-consumer operands pipeline instead of serializing on dscnt 0).
// Each query row is scanned by 2 waves (complementary candidate halves);
// their partial top-8 lists are merged per row at the end (stable by index).
// NOTE: panel-to-panel LDS reuse relies on nK = F/32 being even (true for
// F=256 and F=2048): the last-read buffer is 1, next panel stages buffer 0.
// ---------------------------------------------------------------------------
__global__ __launch_bounds__(256)
void knn_kernel(const _Float16* __restrict__ X, const float* __restrict__ sq,
                int* __restrict__ outIdx, int N, int F)
{
    __shared__ _Float16 sQ[2][128 * STR];
    __shared__ _Float16 sC[2][128 * STR];
    __shared__ float    sD[8][32 * 64];  // per-wave 32 rows x 64 cols distances
    __shared__ float    sSq[2][128];     // candidate sq-norms for current panel
    __shared__ float    sMD[128][16];    // merge scratch: 2 lists x 8 dists/row
    __shared__ int      sMI[128][16];

    const int t = threadIdx.x;
    const int lane = t & 31;
    const int w = t >> 5;
    const int hr = lane & 15, hs = lane >> 4;
    const int qblk = blockIdx.x * 128;
    const int qt0 = (w & 3) * 2;        // first of this wave's 2 query tiles
    const int ch  = w >> 2;             // candidate half (0 or 1)
    const int ct0 = ch * 4;             // first of this wave's 4 candidate tiles
    const int lrow = t >> 2;
    const int lcol = (t & 3) * 8;

    float bestD[8];
    int   bestI[8];
#pragma unroll
    for (int s = 0; s < 8; ++s) { bestD[s] = 3.4e38f; bestI[s] = 0; }
    const int   qrel = qt0 * 16 + lane;          // 0..127: row this lane owns
    const float sqq  = sq[qblk + qrel];

    const int nK = F >> 5;
    for (int j0 = 0; j0 < N; j0 += 128) {
        const int pb = (j0 >> 7) & 1;            // panel parity for sSq buffer
        v8f acc[2][4];
#pragma unroll
        for (int qi = 0; qi < 2; ++qi)
#pragma unroll
            for (int ci = 0; ci < 4; ++ci) acc[qi][ci] = (v8f)0.0f;

        auto stage = [&](int buf, int k0) {
#pragma unroll
            for (int p = 0; p < 2; ++p) {
                int r = lrow + p * 64;
                async_copy_b128(X + (size_t)(qblk + r) * F + k0 + lcol, &sQ[buf][r * STR + lcol]);
                async_copy_b128(X + (size_t)(j0  + r) * F + k0 + lcol, &sC[buf][r * STR + lcol]);
            }
        };

        stage(0, 0);
        if (t < 128) sSq[pb][t] = sq[j0 + t];    // covered by first K-step barrier

        for (int kt = 0; kt < nK; ++kt) {
            wait_async0();
            __syncthreads();
            if (kt + 1 < nK) stage((kt + 1) & 1, (kt + 1) << 5);

            const int buf = kt & 1;
            v16h afrag[2], bfrag[4];
#pragma unroll
            for (int qi = 0; qi < 2; ++qi)
                afrag[qi] = load_frag(&sQ[buf][((qt0 + qi) * 16) * STR], hr, hs);
#pragma unroll
            for (int ci = 0; ci < 4; ++ci)
                bfrag[ci] = load_frag(&sC[buf][((ct0 + ci) * 16) * STR], hr, hs);

#pragma unroll
            for (int qi = 0; qi < 2; ++qi)
#pragma unroll
                for (int ci = 0; ci < 4; ++ci)
                    acc[qi][ci] = __builtin_amdgcn_wmma_f32_16x16x32_f16(
                        false, afrag[qi], false, bfrag[ci],
                        (short)0, acc[qi][ci], false, false);
        }

        // Dump this wave's 32x64 distance block (wave-local; no block barrier).
#pragma unroll
        for (int qi = 0; qi < 2; ++qi)
#pragma unroll
            for (int ci = 0; ci < 4; ++ci)
#pragma unroll
                for (int v = 0; v < 8; ++v)
                    sD[w][(qi * 16 + hs * 8 + v) * 64 + ci * 16 + hr] = acc[qi][ci][v];
        wait_ds0();   // own stores complete; DS ops are in-order per wave

        // All 32 lanes: one query row each, scan 64 candidates (j increasing).
        const float* drow = &sD[w][lane * 64];
        for (int c = 0; c < 64; ++c) {
            int j = j0 + ct0 * 16 + c;
            float d = sqq + sSq[pb][ct0 * 16 + c] - 2.0f * drow[c];
            if (d < bestD[7]) {
                int p = 7;
                while (p > 0 && bestD[p - 1] > d) {
                    bestD[p] = bestD[p - 1]; bestI[p] = bestI[p - 1]; --p;
                }
                bestD[p] = d; bestI[p] = j;
            }
        }
    }

    // Merge the two partial top-8 lists per query row (waves w and w+4 share
    // rows but scanned complementary candidate halves).
#pragma unroll
    for (int s = 0; s < 8; ++s) {
        sMD[qrel][ch * 8 + s] = bestD[s];
        sMI[qrel][ch * 8 + s] = bestI[s];
    }
    __syncthreads();
    if (t < 128) {
        const float* d0 = &sMD[t][0];  const int* i0 = &sMI[t][0];
        const float* d1 = &sMD[t][8];  const int* i1 = &sMI[t][8];
        int a = 0, b = 0;
        int* dst = outIdx + (size_t)(qblk + t) * 8;
#pragma unroll
        for (int s = 0; s < 8; ++s) {
            float da = d0[a], db = d1[b];
            int   ia = i0[a], ib = i1[b];
            bool takeA = (da < db) || (da == db && ia < ib);  // stable by index
            dst[s] = takeA ? ia : ib;
            if (takeA) ++a; else ++b;
        }
    }
}

// ---------------------------------------------------------------------------
// Helper kernels
// ---------------------------------------------------------------------------
__global__ void cvt_f32_f16(const float* __restrict__ in, _Float16* __restrict__ out, int n)
{
    int i = blockIdx.x * blockDim.x + threadIdx.x;
    if (i < n) out[i] = (_Float16)in[i];
}

// out(N x K, f16) = transpose(in(K x N, f32)); 32x32 tiles, 256 threads.
__global__ void cvt_transpose_f16(const float* __restrict__ in, _Float16* __restrict__ out,
                                  int K, int N)
{
    __shared__ float tile[32][33];
    int k0 = blockIdx.y * 32, n0 = blockIdx.x * 32;
    int tx = threadIdx.x & 31, ty = threadIdx.x >> 5;   // 32 x 8
#pragma unroll
    for (int i = ty; i < 32; i += 8)
        tile[i][tx] = in[(size_t)(k0 + i) * N + (n0 + tx)];
    __syncthreads();
#pragma unroll
    for (int i = ty; i < 32; i += 8)
        out[(size_t)(n0 + i) * K + (k0 + tx)] = (_Float16)tile[tx][i];
}

__global__ void sqnorm_kernel(const float* __restrict__ X, float* __restrict__ sq, int N, int F)
{
    int r = blockIdx.x * blockDim.x + threadIdx.x;
    if (r >= N) return;
    const float4* p = (const float4*)(X + (size_t)r * F);
    float s = 0.0f;
    for (int c = 0; c < F / 4; ++c) {
        float4 v = p[c];
        s += v.x * v.x + v.y * v.y + v.z * v.z + v.w * v.w;
    }
    sq[r] = s;
}

// Build edge features for one chunk: E[e] = [h_i , h_j - h_i]  (f16), 2048 edges
__global__ void gather_edges(const float* __restrict__ H, const int* __restrict__ idx,
                             _Float16* __restrict__ E, int n0, int F)
{
    int e = blockIdx.x;
    int node = n0 + (e >> 3);
    int j = idx[node * 8 + (e & 7)];
    const float* hi = H + (size_t)node * F;
    const float* hj = H + (size_t)j * F;
    _Float16* out = E + (size_t)e * (2 * F);
    for (int c = threadIdx.x; c < F; c += blockDim.x) {
        float a = hi[c], b = hj[c];
        out[c]     = (_Float16)a;
        out[F + c] = (_Float16)(b - a);
    }
}

// out[n0+m] = max over 8 neighbor-edges of H2 chunk
__global__ void max_reduce(const float* __restrict__ H2, float* __restrict__ Of,
                           _Float16* __restrict__ Oh, int n0, int C)
{
    int m = blockIdx.y;
    int c = blockIdx.x * blockDim.x + threadIdx.x;
    if (c >= C) return;
    const float* p = H2 + (size_t)(m * 8) * C + c;
    float v = p[0];
#pragma unroll
    for (int s = 1; s < 8; ++s) v = fmaxf(v, p[(size_t)s * C]);
    size_t o = (size_t)(n0 + m) * C + c;
    Of[o] = v;
    if (Oh) Oh[o] = (_Float16)v;
}

// ---------------------------------------------------------------------------
// Launch
// ---------------------------------------------------------------------------
static inline size_t align_up(size_t x) { return (x + 255) & ~(size_t)255; }

extern "C" void kernel_launch(void* const* d_in, const int* in_sizes, int n_in,
                              void* d_out, int out_size, void* d_ws, size_t ws_size,
                              hipStream_t stream)
{
    (void)in_sizes; (void)n_in; (void)out_size; (void)ws_size;
    const int N = 8192;

    const float* x   = (const float*)d_in[0];
    const float* W1a = (const float*)d_in[1];  const float* b1a = (const float*)d_in[2];
    const float* W1b = (const float*)d_in[3];  const float* b1b = (const float*)d_in[4];
    const float* We1 = (const float*)d_in[5];  const float* be1 = (const float*)d_in[6];
    const float* We2 = (const float*)d_in[7];  const float* be2 = (const float*)d_in[8];
    const float* Wf1 = (const float*)d_in[9];  const float* bf1 = (const float*)d_in[10];
    const float* Wf2 = (const float*)d_in[11]; const float* bf2 = (const float*)d_in[12];

    char* ws = (char*)d_ws;
    size_t off = 0;
    auto alloc = [&](size_t bytes) -> void* {
        void* p = ws + off; off = align_up(off + bytes); return p;
    };

    _Float16* xh    = (_Float16*)alloc((size_t)N * 64 * 2);
    _Float16* w1a_t = (_Float16*)alloc((size_t)64 * 1024 * 2);
    _Float16* w1b_t = (_Float16*)alloc((size_t)1024 * 256 * 2);
    _Float16* we1_t = (_Float16*)alloc((size_t)512 * 1024 * 2);
    _Float16* we2_t = (_Float16*)alloc((size_t)1024 * 2048 * 2);
    _Float16* wf1_t = (_Float16*)alloc((size_t)4096 * 1024 * 2);
    _Float16* wf2_t = (_Float16*)alloc((size_t)1024 * 512 * 2);
    _Float16* h0_h  = (_Float16*)alloc((size_t)N * 1024 * 2);
    float*    h_f   = (float*)   alloc((size_t)N * 256 * 4);
    _Float16* h_h   = (_Float16*)alloc((size_t)N * 256 * 2);
    float*    sqb   = (float*)   alloc((size_t)N * 4);
    int*      idx1  = (int*)     alloc((size_t)N * 8 * 4);
    int*      idx2  = (int*)     alloc((size_t)N * 8 * 4);
    float*    o1f   = (float*)   alloc((size_t)N * 2048 * 4);
    _Float16* o1h   = (_Float16*)alloc((size_t)N * 2048 * 2);
    _Float16* Ebuf  = (_Float16*)alloc((size_t)2048 * 4096 * 2);
    _Float16* H1h   = (_Float16*)alloc((size_t)2048 * 1024 * 2);
    float*    H2f   = (float*)   alloc((size_t)2048 * 2048 * 4);

    // Convert activations (row-major) and weights (transposed: N x K) to f16.
    cvt_f32_f16<<<(N * 64 + 255) / 256, 256, 0, stream>>>(x, xh, N * 64);
    auto cvtT = [&](const float* srcKN, _Float16* dstNK, int K, int Nn) {
        cvt_transpose_f16<<<dim3(Nn / 32, K / 32), 256, 0, stream>>>(srcKN, dstNK, K, Nn);
    };
    cvtT(W1a, w1a_t, 64, 1024);
    cvtT(W1b, w1b_t, 1024, 256);
    cvtT(We1, we1_t, 512, 1024);
    cvtT(We2, we2_t, 1024, 2048);
    cvtT(Wf1, wf1_t, 4096, 1024);
    cvtT(Wf2, wf2_t, 1024, 512);

    // --- Point MLP ---
    wmma_gemm<1,0,1><<<dim3(1024/256, N/128), 256, 0, stream>>>(xh,   w1a_t, b1a, nullptr, h0_h, N, 1024, 64);
    wmma_gemm<0,1,1><<<dim3( 256/256, N/128), 256, 0, stream>>>(h0_h, w1b_t, b1b, h_f,    h_h,  N,  256, 1024);

    // --- kNN 1 (F=256) ---
    sqnorm_kernel<<<(N + 255) / 256, 256, 0, stream>>>(h_f, sqb, N, 256);
    knn_kernel<<<N / 128, 256, 0, stream>>>(h_h, sqb, idx1, N, 256);

    // --- EdgeConv 1: 512 -> 1024 (relu) -> 2048, max over 8 nbrs ---
    for (int chunk = 0; chunk < 32; ++chunk) {
        int n0 = chunk * 256;
        gather_edges<<<2048, 256, 0, stream>>>(h_f, idx1, Ebuf, n0, 256);
        wmma_gemm<1,0,1><<<dim3(1024/256, 16), 256, 0, stream>>>(Ebuf, we1_t, be1, nullptr, H1h, 2048, 1024, 512);
        wmma_gemm<0,1,0><<<dim3(2048/256, 16), 256, 0, stream>>>(H1h,  we2_t, be2, H2f, nullptr, 2048, 2048, 1024);
        max_reduce<<<dim3(2048 / 256, 256), 256, 0, stream>>>(H2f, o1f, o1h, n0, 2048);
    }

    // --- kNN 2 (F=2048) ---
    sqnorm_kernel<<<(N + 255) / 256, 256, 0, stream>>>(o1f, sqb, N, 2048);
    knn_kernel<<<N / 128, 256, 0, stream>>>(o1h, sqb, idx2, N, 2048);

    // --- EdgeConv 2: 4096 -> 1024 (relu) -> 512, max over 8 nbrs -> d_out ---
    float* outF = (float*)d_out;
    for (int chunk = 0; chunk < 32; ++chunk) {
        int n0 = chunk * 256;
        gather_edges<<<2048, 256, 0, stream>>>(o1f, idx2, Ebuf, n0, 2048);
        wmma_gemm<1,0,1><<<dim3(1024/256, 16), 256, 0, stream>>>(Ebuf, wf1_t, bf1, nullptr, H1h, 2048, 1024, 4096);
        wmma_gemm<0,1,0><<<dim3( 512/256, 16), 256, 0, stream>>>(H1h,  wf2_t, bf2, H2f, nullptr, 2048, 512, 1024);
        max_reduce<<<dim3(512 / 256, 256), 256, 0, stream>>>(H2f, outF, nullptr, n0, 512);
    }
}
